// MoEBlock_56135222559223
// MI455X (gfx1250) — compile-verified, matrix operand order checked by reading, not run on hardware
//
#include <hip/hip_runtime.h>
#include <cstddef>

typedef __attribute__((ext_vector_type(16))) _Float16 v16h;
typedef __attribute__((ext_vector_type(8)))  float    v8f;

#define TBM 128
#define TBN 128
#define TBK 32
#define TBKP 40    // padded LDS row (40 halves = 80 B: 16B-aligned, bank-rotating)

// ---------------------------------------------------------------------------
// RMSNorm: one block per row.  out = x * rsqrt(mean(x^2)+eps) * (1+scale)
// ---------------------------------------------------------------------------
__global__ __launch_bounds__(256) void rmsnorm_kernel(const float* __restrict__ x,
                                                      const float* __restrict__ scale,
                                                      float* __restrict__ out, int D) {
  __shared__ float red[8];
  long long row = blockIdx.x;
  const float* xr = x + row * (long long)D;
  float s = 0.0f;
  for (int d = threadIdx.x; d < D; d += 256) { float v = xr[d]; s += v * v; }
  for (int off = 16; off > 0; off >>= 1) s += __shfl_xor(s, off, 32);
  if ((threadIdx.x & 31) == 0) red[threadIdx.x >> 5] = s;
  __syncthreads();
  float tot = 0.0f;
#pragma unroll
  for (int w = 0; w < 8; ++w) tot += red[w];
  float inv = rsqrtf(tot / (float)D + 1e-6f);
  float* orow = out + row * (long long)D;
  for (int d = threadIdx.x; d < D; d += 256)
    orow[d] = xr[d] * inv * (1.0f + scale[d]);
}

// ---------------------------------------------------------------------------
// Batched tiled GEMM, fp32 in / fp32 out, f16 WMMA core.
// C[M,N] = alpha * A[M,K] x B[K,N]   (+X | +=C per mode)
// mode 0: C = acc*alpha ; mode 1: C = acc*alpha + X ; mode 2: C += acc*alpha
// Block = 256 threads = 8 wave32 waves in a 2(row)x4(col) grid over a
// 128x128 tile; each wave owns a 64x32 C subtile (4x2 WMMA accumulators).
// A/B tiles staged through LDS as f16 (B transposed for contiguous-k frags);
// LDS rows padded to 80 B to kill bank conflicts on fragment loads.
// ---------------------------------------------------------------------------
__global__ __launch_bounds__(256) void gemm_kernel(
    const float* __restrict__ A, long long strideA, int lda,
    const float* __restrict__ Bm, int ldb,
    float* __restrict__ C, long long strideC, int ldc,
    const float* __restrict__ X, long long strideX,
    int M, int Nn, int Kk, float alpha, int mode) {
  __shared__ _Float16 As[TBM][TBKP];   // 10 KB
  __shared__ _Float16 Bs[TBN][TBKP];   // 10 KB, transposed: [n][k]

  int tid  = threadIdx.x;
  int lane = tid & 31;
  int wave = tid >> 5;
  int wm = wave & 1;      // 2 row groups of 64
  int wn = wave >> 1;     // 4 col groups of 32
  int m0 = blockIdx.y * TBM;
  int n0 = blockIdx.x * TBN;
  int batch = blockIdx.z;
  const float* Ab = A + (long long)batch * strideA;
  float* Cb = C + (long long)batch * strideC;

  v8f acc[4][2];
#pragma unroll
  for (int mt = 0; mt < 4; ++mt)
#pragma unroll
    for (int nt = 0; nt < 2; ++nt)
#pragma unroll
      for (int r = 0; r < 8; ++r) acc[mt][nt][r] = 0.0f;

  int ar = tid >> 1;            // 0..127
  int ac = (tid & 1) * 16;      // 0,16
  int br = tid >> 3;            // 0..31
  int bc = (tid & 7) * 16;      // 0..112
  int fm = lane & 15;
  int fb = (lane >> 4) << 3;

  for (int k0 = 0; k0 < Kk; k0 += TBK) {
    // stage A tile (16 contiguous floats per thread -> contiguous f16 in LDS)
    const float* ap = Ab + (long long)(m0 + ar) * lda + k0 + ac;
#pragma unroll
    for (int j = 0; j < 4; ++j) {
      float4 a = *(const float4*)(ap + 4 * j);
      As[ar][ac + 4 * j + 0] = (_Float16)a.x;
      As[ar][ac + 4 * j + 1] = (_Float16)a.y;
      As[ar][ac + 4 * j + 2] = (_Float16)a.z;
      As[ar][ac + 4 * j + 3] = (_Float16)a.w;
    }
    // stage B tile transposed
    const float* bp = Bm + (long long)(k0 + br) * ldb + n0 + bc;
#pragma unroll
    for (int j = 0; j < 4; ++j) {
      float4 b = *(const float4*)(bp + 4 * j);
      Bs[bc + 4 * j + 0][br] = (_Float16)b.x;
      Bs[bc + 4 * j + 1][br] = (_Float16)b.y;
      Bs[bc + 4 * j + 2][br] = (_Float16)b.z;
      Bs[bc + 4 * j + 3][br] = (_Float16)b.w;
    }
    if (k0 + TBK < Kk) {  // global_prefetch_b8, near-cache hint
      __builtin_prefetch(ap + TBK, 0, 3);
      __builtin_prefetch(bp + (long long)TBK * ldb, 0, 3);
    }
    __syncthreads();

    v16h af[4], bf[2];
#pragma unroll
    for (int mt = 0; mt < 4; ++mt)
#pragma unroll
      for (int i = 0; i < 16; ++i) {
        int kk = fb + (i < 8 ? i : i + 8);
        af[mt][i] = As[wm * 64 + mt * 16 + fm][kk];
      }
#pragma unroll
    for (int nt = 0; nt < 2; ++nt)
#pragma unroll
      for (int i = 0; i < 16; ++i) {
        int kk = fb + (i < 8 ? i : i + 8);
        bf[nt][i] = Bs[wn * 32 + nt * 16 + fm][kk];
      }
#pragma unroll
    for (int mt = 0; mt < 4; ++mt)
#pragma unroll
      for (int nt = 0; nt < 2; ++nt)
        acc[mt][nt] = __builtin_amdgcn_wmma_f32_16x16x32_f16(
            false, af[mt], false, bf[nt], (short)0, acc[mt][nt], false, false);
    __syncthreads();
  }

  const float* Xb = (mode == 1) ? (X + (long long)batch * strideX) : nullptr;
  int hi = (lane >> 4) << 3;
#pragma unroll
  for (int mt = 0; mt < 4; ++mt)
#pragma unroll
    for (int r = 0; r < 8; ++r) {
      int row = m0 + wm * 64 + mt * 16 + r + hi;
#pragma unroll
      for (int nt = 0; nt < 2; ++nt) {
        int col = n0 + wn * 32 + nt * 16 + fm;
        long long i0 = (long long)row * ldc + col;
        float v0 = acc[mt][nt][r] * alpha;
        if (mode == 1) v0 += Xb[i0];
        else if (mode == 2) v0 += Cb[i0];
        Cb[i0] = v0;
      }
    }
}

// ---------------------------------------------------------------------------
// RoPE in place over [B,S,heads,256]; position == sequence index.
// ---------------------------------------------------------------------------
__global__ void rope_kernel(float* __restrict__ x, int S_, int heads, long long total) {
  long long idx = blockIdx.x * 256LL + threadIdx.x;
  if (idx >= total) return;
  int h = (int)(idx & 127);
  long long rest = idx >> 7;                 // (b*S + t)*heads + n
  int t = (int)((rest / heads) % S_);
  float timescale = powf(10000.0f, (float)h * (1.0f / 128.0f));
  float rad = (float)t / timescale;
  float sn, cs;
  sincosf(rad, &sn, &cs);
  float* p = x + rest * 256;
  float x1 = p[h], x2 = p[h + 128];
  p[h]       = x1 * cs - x2 * sn;
  p[h + 128] = x2 * cs + x1 * sn;
}

// ---------------------------------------------------------------------------
// Flash-style MQA attention. 4 waves per block; each wave owns 16 queries of a
// 64-query tile for one (b, head); the 32-key x 256 K/V chunk is staged to LDS
// once per block and shared by all 4 waves (amortizes global K/V traffic 4x).
// LDS rows padded (264 halves = 528 B) so the 16 lanes of a fragment load hit
// 16 distinct banks instead of a single one.
// q:[B,S,8,256] (pre-scaled+roped), k/v:[B,S,256], attn out:[B,S,8,256].
// Softcap tanh(l/50)*50, mask, online softmax, WMMA for QK^T and PV.
// ---------------------------------------------------------------------------
#define HKP 264
#define PSP 40
__global__ __launch_bounds__(128) void attn_kernel(
    const float* __restrict__ q, const float* __restrict__ k, const float* __restrict__ v,
    const unsigned char* __restrict__ mask, float* __restrict__ attn) {
  const int S_ = 2048, Nh = 8;
  __shared__ _Float16 Ks[32][HKP];      // ~16.5 KB
  __shared__ _Float16 Vs[32][HKP];      // ~16.5 KB
  __shared__ _Float16 Ps[4][16][PSP];   // per-wave P re-layout slice, 5 KB

  int tid  = threadIdx.x;
  int lane = tid & 31;
  int wave = tid >> 5;
  int t0 = blockIdx.x * 64 + wave * 16;
  int n  = blockIdx.y;
  int b  = blockIdx.z;
  int fm = lane & 15;
  int fb = (lane >> 4) << 3;
  int mh = (lane >> 4) << 3;   // row offset for C-layout elements

  // Q fragments for all 8 k-steps (H=256), kept in VGPRs, reused for all keys.
  v16h qf[8];
  {
    const float* qrow = q + (((size_t)b * S_ + (t0 + fm)) * Nh + n) * 256;
#pragma unroll
    for (int kk = 0; kk < 8; ++kk)
#pragma unroll
      for (int i = 0; i < 16; ++i) {
        int h = kk * 32 + fb + (i < 8 ? i : i + 8);
        qf[kk][i] = (_Float16)qrow[h];
      }
  }

  v8f O[16];
#pragma unroll
  for (int hc = 0; hc < 16; ++hc)
#pragma unroll
    for (int r = 0; r < 8; ++r) O[hc][r] = 0.0f;

  float row_max[8], row_sum[8];
#pragma unroll
  for (int r = 0; r < 8; ++r) { row_max[r] = -3.0e38f; row_sum[r] = 0.0f; }

  for (int s0 = 0; s0 < S_; s0 += 32) {
    // cooperative K/V staging: each thread handles 8 contiguous h of one row
#pragma unroll
    for (int it = 0; it < 8; ++it) {
      int idx = tid + it * 128;          // 0..1023
      int r  = idx >> 5;                 // 0..31
      int c  = (idx & 31) * 8;           // 0..248
      const float* krow = k + ((size_t)b * S_ + s0 + r) * 256 + c;
      const float* vrow = v + ((size_t)b * S_ + s0 + r) * 256 + c;
      float4 k0 = *(const float4*)(krow);
      float4 k1 = *(const float4*)(krow + 4);
      float4 v0 = *(const float4*)(vrow);
      float4 v1 = *(const float4*)(vrow + 4);
      Ks[r][c + 0] = (_Float16)k0.x; Ks[r][c + 1] = (_Float16)k0.y;
      Ks[r][c + 2] = (_Float16)k0.z; Ks[r][c + 3] = (_Float16)k0.w;
      Ks[r][c + 4] = (_Float16)k1.x; Ks[r][c + 5] = (_Float16)k1.y;
      Ks[r][c + 6] = (_Float16)k1.z; Ks[r][c + 7] = (_Float16)k1.w;
      Vs[r][c + 0] = (_Float16)v0.x; Vs[r][c + 1] = (_Float16)v0.y;
      Vs[r][c + 2] = (_Float16)v0.z; Vs[r][c + 3] = (_Float16)v0.w;
      Vs[r][c + 4] = (_Float16)v1.x; Vs[r][c + 5] = (_Float16)v1.y;
      Vs[r][c + 6] = (_Float16)v1.z; Vs[r][c + 7] = (_Float16)v1.w;
    }
    __syncthreads();

    // QK^T logits for this wave's 16x32 tile (two 16x16 subtiles)
    float Lg[2][8];
#pragma unroll
    for (int sub = 0; sub < 2; ++sub) {
      v8f acc;
#pragma unroll
      for (int r = 0; r < 8; ++r) acc[r] = 0.0f;
#pragma unroll
      for (int kk = 0; kk < 8; ++kk) {
        v16h bf;
#pragma unroll
        for (int i = 0; i < 16; ++i) {
          int h = kk * 32 + fb + (i < 8 ? i : i + 8);
          bf[i] = Ks[sub * 16 + fm][h];
        }
        acc = __builtin_amdgcn_wmma_f32_16x16x32_f16(false, qf[kk], false, bf, (short)0, acc, false, false);
      }
#pragma unroll
      for (int r = 0; r < 8; ++r) {
        int t = t0 + r + mh;
        int s = s0 + sub * 16 + fm;
        float l = 50.0f * tanhf(acc[r] * 0.02f);
        if (!mask[((size_t)b * S_ + t) * S_ + s]) l = -2.3819763e38f;
        Lg[sub][r] = l;
      }
    }

    // online softmax (rows live across the 16 lanes of each half-wave group)
    float corr[8];
#pragma unroll
    for (int r = 0; r < 8; ++r) {
      float mx = fmaxf(Lg[0][r], Lg[1][r]);
      mx = fmaxf(mx, __shfl_xor(mx, 1, 16));
      mx = fmaxf(mx, __shfl_xor(mx, 2, 16));
      mx = fmaxf(mx, __shfl_xor(mx, 4, 16));
      mx = fmaxf(mx, __shfl_xor(mx, 8, 16));
      float nm = fmaxf(row_max[r], mx);
      corr[r] = expf(row_max[r] - nm);
      row_max[r] = nm;
      float p0 = expf(Lg[0][r] - nm);
      float p1 = expf(Lg[1][r] - nm);
      Lg[0][r] = p0; Lg[1][r] = p1;
      float ps = p0 + p1;
      ps += __shfl_xor(ps, 1, 16);
      ps += __shfl_xor(ps, 2, 16);
      ps += __shfl_xor(ps, 4, 16);
      ps += __shfl_xor(ps, 8, 16);
      row_sum[r] = row_sum[r] * corr[r] + ps;
    }

    // restage P from C-layout into A-fragment layout via this wave's LDS slice
#pragma unroll
    for (int r = 0; r < 8; ++r) {
      int ml = r + mh;
      Ps[wave][ml][fm]      = (_Float16)Lg[0][r];
      Ps[wave][ml][16 + fm] = (_Float16)Lg[1][r];
    }
    v16h pf;
#pragma unroll
    for (int i = 0; i < 16; ++i) {
      int kx = fb + (i < 8 ? i : i + 8);
      pf[i] = Ps[wave][fm][kx];
    }

    // PV accumulate over all 16 h-chunks
#pragma unroll
    for (int hc = 0; hc < 16; ++hc) {
#pragma unroll
      for (int r = 0; r < 8; ++r) O[hc][r] *= corr[r];
      v16h vf;
#pragma unroll
      for (int i = 0; i < 16; ++i) {
        int sl = fb + (i < 8 ? i : i + 8);
        vf[i] = Vs[sl][hc * 16 + fm];
      }
      O[hc] = __builtin_amdgcn_wmma_f32_16x16x32_f16(false, pf, false, vf, (short)0, O[hc], false, false);
    }
    __syncthreads();   // protect Ks/Vs before next chunk restage
  }

  // normalize + store
#pragma unroll
  for (int hc = 0; hc < 16; ++hc)
#pragma unroll
    for (int r = 0; r < 8; ++r) {
      int t = t0 + r + mh;
      int h = hc * 16 + fm;
      attn[(((size_t)b * S_ + t) * Nh + n) * 256 + h] = O[hc][r] / row_sum[r];
    }
}

// ---------------------------------------------------------------------------
// g = gelu_tanh(g) * u
// ---------------------------------------------------------------------------
__global__ void gelumul_kernel(float* __restrict__ g, const float* __restrict__ u, long long n) {
  long long i = blockIdx.x * 256LL + threadIdx.x;
  if (i >= n) return;
  float x = g[i];
  float t = tanhf(0.7978845608f * (x + 0.044715f * x * x * x));
  g[i] = 0.5f * x * (1.0f + t) * u[i];
}

// ---------------------------------------------------------------------------
extern "C" void kernel_launch(void* const* d_in, const int* in_sizes, int n_in,
                              void* d_out, int out_size, void* d_ws, size_t ws_size,
                              hipStream_t stream) {
  (void)in_sizes; (void)n_in; (void)out_size; (void)ws_size;
  constexpr int B_ = 2, L1_ = 1536, L2_ = 512, D1_ = 2048, D2_ = 1024;
  constexpr int F1_ = 16384, F2_ = 4096, N_ = 8, H_ = 256, S_ = 2048, FC = 2048;
  typedef long long ll;

  const float* x_p  = (const float*)d_in[0];
  const float* x_s  = (const float*)d_in[1];
  const unsigned char* mask = (const unsigned char*)d_in[2];
  const float* p_pre_attn = (const float*)d_in[3];
  const float* p_wq = (const float*)d_in[4];
  const float* p_wk = (const float*)d_in[5];
  const float* p_wv = (const float*)d_in[6];
  const float* p_wo = (const float*)d_in[7];
  const float* p_pre_ffw = (const float*)d_in[8];
  const float* p_wgate = (const float*)d_in[9];
  const float* p_wup   = (const float*)d_in[10];
  const float* p_wdown = (const float*)d_in[11];
  const float* s_pre_attn = (const float*)d_in[12];
  const float* s_wq = (const float*)d_in[13];
  const float* s_wk = (const float*)d_in[14];
  const float* s_wv = (const float*)d_in[15];
  const float* s_wo = (const float*)d_in[16];
  const float* s_pre_ffw = (const float*)d_in[17];
  const float* s_wgate = (const float*)d_in[18];
  const float* s_wup   = (const float*)d_in[19];
  const float* s_wdown = (const float*)d_in[20];

  float* out_p = (float*)d_out;
  float* out_s = out_p + (size_t)B_ * L1_ * D1_;

  // workspace layout (floats), ~185 MB total
  float* ws = (float*)d_ws;
  size_t o = 0;
  float* xn_p = ws + o; o += (size_t)B_ * L1_ * D1_;   // reused later as yn_p
  float* xn_s = ws + o; o += (size_t)B_ * L2_ * D2_;   // reused later as yn_s
  float* qb   = ws + o; o += (size_t)B_ * S_ * N_ * H_;
  float* kb   = ws + o; o += (size_t)B_ * S_ * H_;
  float* vb   = ws + o; o += (size_t)B_ * S_ * H_;
  float* ab   = ws + o; o += (size_t)B_ * S_ * N_ * H_;
  float* y0p  = ws + o; o += (size_t)B_ * L1_ * D1_;
  float* y0s  = ws + o; o += (size_t)B_ * L2_ * D2_;
  float* gb   = ws + o; o += (size_t)B_ * L1_ * FC;
  float* ub   = ws + o; o += (size_t)B_ * L1_ * FC;

  const float qscale = 0.0625f;  // 1/sqrt(256)

  // 1) pre-attention RMSNorm
  rmsnorm_kernel<<<B_ * L1_, 256, 0, stream>>>(x_p, p_pre_attn, xn_p, D1_);
  rmsnorm_kernel<<<B_ * L2_, 256, 0, stream>>>(x_s, s_pre_attn, xn_s, D2_);

  // 2) QKV projections (batched over b via grid.z)
  gemm_kernel<<<dim3((N_*H_)/TBN, L1_/TBM, B_), 256, 0, stream>>>(
      xn_p, (ll)L1_*D1_, D1_, p_wq, N_*H_, qb, (ll)S_*N_*H_, N_*H_, nullptr, 0,
      L1_, N_*H_, D1_, qscale, 0);
  gemm_kernel<<<dim3(H_/TBN, L1_/TBM, B_), 256, 0, stream>>>(
      xn_p, (ll)L1_*D1_, D1_, p_wk, H_, kb, (ll)S_*H_, H_, nullptr, 0,
      L1_, H_, D1_, 1.0f, 0);
  gemm_kernel<<<dim3(H_/TBN, L1_/TBM, B_), 256, 0, stream>>>(
      xn_p, (ll)L1_*D1_, D1_, p_wv, H_, vb, (ll)S_*H_, H_, nullptr, 0,
      L1_, H_, D1_, 1.0f, 0);
  gemm_kernel<<<dim3((N_*H_)/TBN, L2_/TBM, B_), 256, 0, stream>>>(
      xn_s, (ll)L2_*D2_, D2_, s_wq, N_*H_, qb + (size_t)L1_*N_*H_, (ll)S_*N_*H_, N_*H_, nullptr, 0,
      L2_, N_*H_, D2_, qscale, 0);
  gemm_kernel<<<dim3(H_/TBN, L2_/TBM, B_), 256, 0, stream>>>(
      xn_s, (ll)L2_*D2_, D2_, s_wk, H_, kb + (size_t)L1_*H_, (ll)S_*H_, H_, nullptr, 0,
      L2_, H_, D2_, 1.0f, 0);
  gemm_kernel<<<dim3(H_/TBN, L2_/TBM, B_), 256, 0, stream>>>(
      xn_s, (ll)L2_*D2_, D2_, s_wv, H_, vb + (size_t)L1_*H_, (ll)S_*H_, H_, nullptr, 0,
      L2_, H_, D2_, 1.0f, 0);

  // 3) RoPE (positions == sequence index after concat)
  {
    long long tq = (long long)B_ * S_ * N_ * 128;
    rope_kernel<<<(unsigned)((tq + 255) / 256), 256, 0, stream>>>(qb, S_, N_, tq);
    long long tk = (long long)B_ * S_ * 1 * 128;
    rope_kernel<<<(unsigned)((tk + 255) / 256), 256, 0, stream>>>(kb, S_, 1, tk);
  }

  // 4) attention (64 queries per block, 4 waves)
  attn_kernel<<<dim3(S_ / 64, N_, B_), 128, 0, stream>>>(qb, kb, vb, mask, ab);

  // 5) output projection + residual
  gemm_kernel<<<dim3(D1_/TBN, L1_/TBM, B_), 256, 0, stream>>>(
      ab, (ll)S_*N_*H_, N_*H_, p_wo, D1_, y0p, (ll)L1_*D1_, D1_, x_p, (ll)L1_*D1_,
      L1_, D1_, N_*H_, 1.0f, 1);
  gemm_kernel<<<dim3(D2_/TBN, L2_/TBM, B_), 256, 0, stream>>>(
      ab + (size_t)L1_*N_*H_, (ll)S_*N_*H_, N_*H_, s_wo, D2_, y0s, (ll)L2_*D2_, D2_, x_s, (ll)L2_*D2_,
      L2_, D2_, N_*H_, 1.0f, 1);

  // 6) pre-FFN RMSNorm (reuse xn buffers)
  rmsnorm_kernel<<<B_ * L1_, 256, 0, stream>>>(y0p, p_pre_ffw, xn_p, D1_);
  rmsnorm_kernel<<<B_ * L2_, 256, 0, stream>>>(y0s, s_pre_ffw, xn_s, D2_);

  // 7) FFN prefix, chunked over F1 (bounded workspace; sequential launches
  //    make the += down-projection race-free)
  {
    const int Mp = B_ * L1_;                    // 3072, weights shared across b
    for (int fc = 0; fc < F1_; fc += FC) {
      gemm_kernel<<<dim3(FC/TBN, Mp/TBM, 1), 256, 0, stream>>>(
          xn_p, 0, D1_, p_wgate + fc, F1_, gb, 0, FC, nullptr, 0, Mp, FC, D1_, 1.0f, 0);
      gemm_kernel<<<dim3(FC/TBN, Mp/TBM, 1), 256, 0, stream>>>(
          xn_p, 0, D1_, p_wup + fc, F1_, ub, 0, FC, nullptr, 0, Mp, FC, D1_, 1.0f, 0);
      long long ng = (long long)Mp * FC;
      gelumul_kernel<<<(unsigned)((ng + 255) / 256), 256, 0, stream>>>(gb, ub, ng);
      gemm_kernel<<<dim3(D1_/TBN, Mp/TBM, 1), 256, 0, stream>>>(
          gb, 0, FC, p_wdown + (size_t)fc * D1_, D1_, out_p, 0, D1_,
          (fc == 0 ? x_p : nullptr), 0, Mp, D1_, FC, 1.0f, (fc == 0 ? 1 : 2));
    }
  }

  // 8) FFN suffix
  {
    const int Ms = B_ * L2_;                    // 1024
    for (int fc = 0; fc < F2_; fc += FC) {
      gemm_kernel<<<dim3(FC/TBN, Ms/TBM, 1), 256, 0, stream>>>(
          xn_s, 0, D2_, s_wgate + fc, F2_, gb, 0, FC, nullptr, 0, Ms, FC, D2_, 1.0f, 0);
      gemm_kernel<<<dim3(FC/TBN, Ms/TBM, 1), 256, 0, stream>>>(
          xn_s, 0, D2_, s_wup + fc, F2_, ub, 0, FC, nullptr, 0, Ms, FC, D2_, 1.0f, 0);
      long long ng = (long long)Ms * FC;
      gelumul_kernel<<<(unsigned)((ng + 255) / 256), 256, 0, stream>>>(gb, ub, ng);
      gemm_kernel<<<dim3(D2_/TBN, Ms/TBM, 1), 256, 0, stream>>>(
          gb, 0, FC, s_wdown + (size_t)fc * D2_, D2_, out_s, 0, D2_,
          (fc == 0 ? x_s : nullptr), 0, Ms, D2_, FC, 1.0f, (fc == 0 ? 1 : 2));
    }
  }
}
